// LocalAttention_15908558865361
// MI455X (gfx1250) — matrix operand order repaired and checked by reading, not compiled, not run on hardware
//
#include <hip/hip_runtime.h>
#include <math.h>

typedef __bf16 bf16t;
typedef __attribute__((ext_vector_type(2)))  float  v2f;
typedef __attribute__((ext_vector_type(8)))  float  v8f;
typedef __attribute__((ext_vector_type(4)))  __bf16 v4bf;
typedef __attribute__((ext_vector_type(8)))  __bf16 v8bf;
typedef __attribute__((ext_vector_type(16))) __bf16 v16bf;

// fp32 WMMA: D = A(16x4) * B(4x16) + C
#define WMMA_F32(a, b, c) \
    __builtin_amdgcn_wmma_f32_16x16x4_f32(false, (a), false, (b), (short)0, (c), false, false)
// bf16 WMMA: D = A(16x32) * B(32x16) + C, f32 accumulate
#define WMMA_BF16(a, b, c) \
    __builtin_amdgcn_wmma_f32_16x16x32_bf16(false, (a), false, (b), (short)0, (c), false, false)

struct bfsplit { bf16t h, l; };
__device__ __forceinline__ bfsplit split_bf16(float x) {
    bf16t h = (bf16t)x;                       // RNE to bf16
    bf16t l = (bf16t)(x - (float)h);          // residual: ~8 more mantissa bits
    return {h, l};
}
__device__ __forceinline__ void split4(float4 v, v4bf& h, v4bf& l) {
    bfsplit s0 = split_bf16(v.x), s1 = split_bf16(v.y),
            s2 = split_bf16(v.z), s3 = split_bf16(v.w);
    h = (v4bf){s0.h, s1.h, s2.h, s3.h};
    l = (v4bf){s0.l, s1.l, s2.l, s3.l};
}

// -------------------------------------------------------------------------
// C[M,N] = A[M,K] @ Bw[N,K]^T + bias[N]   -- bf16x3 split, f32-accurate
// 64x64 block tile, BK=64, 8 waves: wave -> 16x32 sub-tile.
// -------------------------------------------------------------------------
__global__ __launch_bounds__(256) void gemm_bias_64x64_bf16x3(
    const float* __restrict__ A, const float* __restrict__ Bw,
    const float* __restrict__ bias, float* __restrict__ C,
    int M, int N, int K)
{
    constexpr int BK  = 64;
    constexpr int LDK = BK + 8;                 // 72 bf16 = 144B row stride (16B aligned)
    __shared__ bf16t Ah[64][LDK], Al[64][LDK];  // 4 * 64*72*2 = 36,864 B total
    __shared__ bf16t Bh[64][LDK], Bl[64][LDK];

    const int tid  = threadIdx.x;
    const int wave = tid >> 5;
    const int lane = tid & 31;
    const int l16  = lane & 15;
    const int half = lane >> 4;

    const int bm = blockIdx.y * 64;
    const int bn = blockIdx.x * 64;
    const int wm = (wave >> 1) * 16;   // 0,16,32,48
    const int wn = (wave & 1) * 32;    // 0,32

    v8f acc0 = {};
    v8f acc1 = {};

    for (int k0 = 0; k0 < K; k0 += BK) {
        // stage 64xBK fp32 of A and Bw, split into hi/lo bf16 (4 float4 each/thread)
        #pragma unroll
        for (int i = 0; i < 4; ++i) {
            int idx = tid + i * 256;       // 0..1023
            int r   = idx >> 4;            // 0..63
            int c   = (idx & 15) * 4;      // 0..60
            float4 av = *(const float4*)&A [(size_t)(bm + r) * K + k0 + c];
            float4 bv = *(const float4*)&Bw[(size_t)(bn + r) * K + k0 + c];
            v4bf ah, al, bh, bl;
            split4(av, ah, al);
            split4(bv, bh, bl);
            *(v4bf*)&Ah[r][c] = ah;  *(v4bf*)&Al[r][c] = al;
            *(v4bf*)&Bh[r][c] = bh;  *(v4bf*)&Bl[r][c] = bl;
        }
        __syncthreads();

        #pragma unroll
        for (int kk = 0; kk < BK; kk += 32) {
            // A fragment (16x32 bf16): elems 0-7 -> K=kk+half*8+e, elems 8-15 -> +16
            const int ka = kk + half * 8;
            v8bf a0h = *(const v8bf*)&Ah[wm + l16][ka];
            v8bf a1h = *(const v8bf*)&Ah[wm + l16][ka + 16];
            v8bf a0l = *(const v8bf*)&Al[wm + l16][ka];
            v8bf a1l = *(const v8bf*)&Al[wm + l16][ka + 16];
            v16bf a_hi = __builtin_shufflevector(a0h, a1h, 0,1,2,3,4,5,6,7,8,9,10,11,12,13,14,15);
            v16bf a_lo = __builtin_shufflevector(a0l, a1l, 0,1,2,3,4,5,6,7,8,9,10,11,12,13,14,15);

            // B fragment (32x16 bf16): lane = column, elems 0-7 -> K=kk+half*8+e, elems 8-15 -> +16
            const int kb = kk + half * 8;
            {
                const int n = wn + l16;
                v16bf b_hi = __builtin_shufflevector(*(const v8bf*)&Bh[n][kb], *(const v8bf*)&Bh[n][kb + 16], 0,1,2,3,4,5,6,7,8,9,10,11,12,13,14,15);
                v16bf b_lo = __builtin_shufflevector(*(const v8bf*)&Bl[n][kb], *(const v8bf*)&Bl[n][kb + 16], 0,1,2,3,4,5,6,7,8,9,10,11,12,13,14,15);
                acc0 = WMMA_BF16(a_hi, b_hi, acc0);
                acc0 = WMMA_BF16(a_lo, b_hi, acc0);
                acc0 = WMMA_BF16(a_hi, b_lo, acc0);
            }
            {
                const int n = wn + 16 + l16;
                v16bf b_hi = __builtin_shufflevector(*(const v8bf*)&Bh[n][kb], *(const v8bf*)&Bh[n][kb + 16], 0,1,2,3,4,5,6,7,8,9,10,11,12,13,14,15);
                v16bf b_lo = __builtin_shufflevector(*(const v8bf*)&Bl[n][kb], *(const v8bf*)&Bl[n][kb + 16], 0,1,2,3,4,5,6,7,8,9,10,11,12,13,14,15);
                acc1 = WMMA_BF16(a_hi, b_hi, acc1);
                acc1 = WMMA_BF16(a_lo, b_hi, acc1);
                acc1 = WMMA_BF16(a_hi, b_lo, acc1);
            }
        }
        __syncthreads();
    }

    // C/D layout (f32 16x16): VGPR r -> row = r + 8*half ; lane%16 -> col
    const int   row0 = bm + wm + half * 8;
    const float bv0  = bias[bn + wn + l16];
    const float bv1  = bias[bn + wn + 16 + l16];
    #pragma unroll
    for (int r = 0; r < 8; ++r) {
        C[(size_t)(row0 + r) * N + bn + wn + l16]      = acc0[r] + bv0;
        C[(size_t)(row0 + r) * N + bn + wn + 16 + l16] = acc1[r] + bv1;
    }
}

// -------------------------------------------------------------------------
// One workgroup per (b, h, window): causal softmax(q k^T / 8) v, W=128, dh=64
// qkv: [B, L, 3*1024] fp32 ; aout: [B, L, 1024] fp32   (exact fp32 WMMA path)
// -------------------------------------------------------------------------
__global__ __launch_bounds__(256) void local_attn_128(
    const float* __restrict__ qkv, float* __restrict__ aout)
{
    constexpr int L = 4096, D3 = 3072, Dm = 1024;

    union Smem {
        struct { float q[128][68]; float k[128][68]; } p1;   // 69,632 B
        struct { float p[128][132]; float vT[64][132]; } p2; // 101,376 B
    };
    __shared__ Smem sm;

    const int blk = blockIdx.x;       // 0..2047
    const int b   = blk >> 9;         // / (16*32)
    const int h   = (blk >> 5) & 15;
    const int win = blk & 31;

    const int tid  = threadIdx.x;
    const int wave = tid >> 5;
    const int lane = tid & 31;
    const int l16  = lane & 15;
    const int half = lane >> 4;

    const size_t rowbase = ((size_t)b * L + (size_t)win * 128) * D3 + (size_t)h * 64;

    // ---- stage q, k ----
    #pragma unroll
    for (int i = 0; i < 8; ++i) {
        int idx = tid + i * 256;      // 0..2047
        int r   = idx >> 4;           // 0..127
        int c   = (idx & 15) * 4;     // 0..60
        *(float4*)&sm.p1.q[r][c] = *(const float4*)&qkv[rowbase + (size_t)r * D3 + c];
        *(float4*)&sm.p1.k[r][c] = *(const float4*)&qkv[rowbase + (size_t)r * D3 + 1024 + c];
    }
    __syncthreads();

    // ---- scores: wave owns rows [wave*16, wave*16+16) ----
    const int wm = wave * 16;
    v8f sc[8] = {};
    for (int kk = 0; kk < 64; kk += 4) {
        const int kb = kk + half * 2;
        v2f a = *(const v2f*)&sm.p1.q[wm + l16][kb];
        #pragma unroll
        for (int nt = 0; nt < 8; ++nt) {
            v2f bf = *(const v2f*)&sm.p1.k[nt * 16 + l16][kb];
            sc[nt] = WMMA_F32(a, bf, sc[nt]);
        }
    }
    __syncthreads();   // q/k reads done; LDS may be re-used

    // ---- scale + causal mask + row softmax in registers ----
    const float scale = 0.125f;       // 1/sqrt(64)
    float rmax[8], rsum[8];
    #pragma unroll
    for (int r = 0; r < 8; ++r) rmax[r] = -1e30f;

    #pragma unroll
    for (int nt = 0; nt < 8; ++nt) {
        const int col = nt * 16 + l16;
        #pragma unroll
        for (int r = 0; r < 8; ++r) {
            const int row = wm + half * 8 + r;
            float s = sc[nt][r] * scale;
            s = (col <= row) ? s : -INFINITY;
            sc[nt][r] = s;
            rmax[r] = fmaxf(rmax[r], s);
        }
    }
    #pragma unroll
    for (int r = 0; r < 8; ++r) {     // reduce across the 16-lane half
        float m = rmax[r];
        m = fmaxf(m, __shfl_xor(m, 1));
        m = fmaxf(m, __shfl_xor(m, 2));
        m = fmaxf(m, __shfl_xor(m, 4));
        m = fmaxf(m, __shfl_xor(m, 8));
        rmax[r] = m;
        rsum[r] = 0.0f;
    }
    #pragma unroll
    for (int nt = 0; nt < 8; ++nt)
        #pragma unroll
        for (int r = 0; r < 8; ++r) {
            float p = __expf(sc[nt][r] - rmax[r]);
            sc[nt][r] = p;
            rsum[r] += p;
        }
    #pragma unroll
    for (int r = 0; r < 8; ++r) {
        float s = rsum[r];
        s += __shfl_xor(s, 1);
        s += __shfl_xor(s, 2);
        s += __shfl_xor(s, 4);
        s += __shfl_xor(s, 8);
        rsum[r] = 1.0f / s;
    }

    // write normalized probabilities to LDS (A-matrix for the PV GEMM)
    #pragma unroll
    for (int nt = 0; nt < 8; ++nt)
        #pragma unroll
        for (int r = 0; r < 8; ++r)
            sm.p2.p[wm + half * 8 + r][nt * 16 + l16] = sc[nt][r] * rsum[r];

    // stage v transposed: vT[d][krow]
    #pragma unroll
    for (int i = 0; i < 32; ++i) {
        int idx = tid + i * 256;      // 0..8191
        int r   = idx >> 6;           // key row 0..127
        int d   = idx & 63;
        sm.p2.vT[d][r] = qkv[rowbase + (size_t)r * D3 + 2048 + d];
    }
    __syncthreads();

    // ---- out = P @ V : wave strip 16 rows x 64 cols ----
    v8f o[4] = {};
    for (int kk = 0; kk < 128; kk += 4) {
        const int kb = kk + half * 2;
        v2f a = *(const v2f*)&sm.p2.p[wm + l16][kb];
        #pragma unroll
        for (int dt = 0; dt < 4; ++dt) {
            v2f bf = *(const v2f*)&sm.p2.vT[dt * 16 + l16][kb];
            o[dt] = WMMA_F32(a, bf, o[dt]);
        }
    }

    const size_t obase = ((size_t)b * L + (size_t)win * 128) * Dm + (size_t)h * 64;
    #pragma unroll
    for (int dt = 0; dt < 4; ++dt)
        #pragma unroll
        for (int r = 0; r < 8; ++r)
            aout[obase + (size_t)(wm + half * 8 + r) * Dm + dt * 16 + l16] = o[dt][r];
}

// -------------------------------------------------------------------------
extern "C" void kernel_launch(void* const* d_in, const int* in_sizes, int n_in,
                              void* d_out, int out_size, void* d_ws, size_t ws_size,
                              hipStream_t stream)
{
    (void)in_sizes; (void)n_in; (void)out_size; (void)ws_size;

    const float* x     = (const float*)d_in[0];   // [4,4096,1024]
    const float* qkv_w = (const float*)d_in[1];   // [3072,1024]
    const float* qkv_b = (const float*)d_in[2];   // [3072]
    const float* out_w = (const float*)d_in[3];   // [1024,1024]
    const float* out_b = (const float*)d_in[4];   // [1024]
    float* out = (float*)d_out;                   // [4,4096,1024]

    const int Mrows = 4 * 4096;   // 16384
    const int Dm = 1024, D3 = 3072;

    float* qkv  = (float*)d_ws;                   // Mrows * 3072 fp32 (192 MB)
    float* aout = qkv + (size_t)Mrows * D3;       // Mrows * 1024 fp32 (64 MB)

    dim3 blk(256);

    // 1) qkv = x @ qkv_w^T + qkv_b   (bf16x3 split WMMA)
    gemm_bias_64x64_bf16x3<<<dim3(D3 / 64, Mrows / 64), blk, 0, stream>>>(
        x, qkv_w, qkv_b, qkv, Mrows, D3, Dm);

    // 2) windowed causal attention per (b, h, window)  (fp32 WMMA)
    local_attn_128<<<dim3(4 * 16 * 32), blk, 0, stream>>>(qkv, aout);

    // 3) out = aout @ out_w^T + out_b  (bf16x3 split WMMA)
    gemm_bias_64x64_bf16x3<<<dim3(Dm / 64, Mrows / 64), blk, 0, stream>>>(
        aout, out_w, out_b, out, Mrows, Dm, Dm);
}
